// VanillaRNN_75239237091384
// MI455X (gfx1250) — matrix operand order repaired and checked by reading, not compiled, not run on hardware
//
#include <hip/hip_runtime.h>

typedef __attribute__((ext_vector_type(16))) _Float16 v16h;
typedef __attribute__((ext_vector_type(8)))  float    v8f;

#define VOCAB  128
#define EMBED  32
#define HIDDEN 128
#define BATCH  256
#define SEQ    1024
#define TILE_M 16   // batch rows per workgroup (one WMMA M-tile)
#define NWAVES 8    // one wave per 16-col N tile (HIDDEN/16 == VOCAB/16 == 8)

// A-operand (16x32 f16) gather from row-major LDS [16 x ld] in contiguous k order (same as B):
// lane<16 : row=lane,    K = kbase + 0..15
// lane>=16: row=lane-16, K = kbase + 16..31
__device__ __forceinline__ v16h load_A_lds(const _Float16* base, int ld, int kbase, int lane) {
    const int row = lane & 15;
    const _Float16* p = base + row * ld + kbase + ((lane < 16) ? 0 : 16);
    v16h a;
#pragma unroll
    for (int i = 0; i < 8; ++i) { a[i] = p[i]; a[8 + i] = p[8 + i]; }
    return a;
}

// B-operand (32x16 f16) gather from row-major global f32 weights [K x ld]:
// lane<16 : N = ncol+lane,    K = kbase + 0..15 (one K per packed half)
// lane>=16: N = ncol+lane-16, K = kbase + 16..31
__device__ __forceinline__ v16h load_B_global_f32(const float* w, int ld, int kbase, int ncol, int lane) {
    const int n  = ncol + (lane & 15);
    const int k0 = kbase + ((lane < 16) ? 0 : 16);
    v16h b;
#pragma unroll
    for (int i = 0; i < 16; ++i) b[i] = (_Float16)w[(k0 + i) * ld + n];
    return b;
}

__global__ __launch_bounds__(TILE_M * NWAVES * 2, 1)
void rnn_seq_kernel(const int* __restrict__ x, const float* __restrict__ embed,
                    const float* __restrict__ W_ih, const float* __restrict__ W_hh,
                    const float* __restrict__ b_h,  const float* __restrict__ W_fc,
                    const float* __restrict__ b_fc, float* __restrict__ out) {
    __shared__ _Float16 sEmb[VOCAB * EMBED];        // 8 KB: f16 embedding table
    __shared__ _Float16 sH[2][TILE_M * HIDDEN];     // 8 KB: ping-pong hidden state (f16)

    const int tid     = threadIdx.x;
    const int lane    = tid & 31;
    const int wid     = tid >> 5;       // 0..7 -> N tile
    const int ncol    = wid * 16;
    const int rowBase = blockIdx.x * TILE_M;

    // Stage embedding table as f16; zero both hidden buffers (h0 = 0).
    for (int i = tid; i < VOCAB * EMBED; i += blockDim.x)
        sEmb[i] = (_Float16)embed[i];
    for (int i = tid; i < 2 * TILE_M * HIDDEN; i += blockDim.x)
        (&sH[0][0])[i] = (_Float16)0.0f;

    // Preload time-invariant per-wave B tiles (weights) into registers, f32->f16.
    v16h Bih = load_B_global_f32(W_ih, HIDDEN, 0, ncol, lane);
    v16h Bhh[4], Bfc[4];
#pragma unroll
    for (int kc = 0; kc < 4; ++kc) {
        Bhh[kc] = load_B_global_f32(W_hh, HIDDEN, kc * 32, ncol, lane);
        Bfc[kc] = load_B_global_f32(W_fc, VOCAB,  kc * 32, ncol, lane);
    }
    // Bias folded into accumulator init; per C-layout every VGPR of a lane shares N.
    const float biasH = b_h[ncol + (lane & 15)];
    const float biasF = b_fc[ncol + (lane & 15)];

    const int xRow  = rowBase + (lane & 15);
    const int mhigh = (lane < 16) ? 0 : 8;

    __syncthreads();

    // Prologue: embedding contribution for t=0 (independent of the recurrence).
    v8f cE;
#pragma unroll
    for (int i = 0; i < 8; ++i) cE[i] = biasH;
    {
        const int idx0 = x[xRow * SEQ];
        v16h aE0 = load_A_lds(&sEmb[idx0 * EMBED], 0, 0, lane);
        cE = __builtin_amdgcn_wmma_f32_16x16x32_f16(false, aE0, false, Bih, (short)0, cE, false, false);
    }

    for (int t = 0; t < SEQ; ++t) {
        const int cur = t & 1, nxt = cur ^ 1;

        // Prefetch next token index early so the global load overlaps the WMMAs.
        const int tn   = (t + 1 < SEQ) ? (t + 1) : t;
        const int idxN = x[xRow * SEQ + tn];

        // ---- hidden update: h_new = tanh(cE + h @ W_hh), cE = e_t@W_ih + b_h ----
        // Issue all A loads first (clause of 8 ds_load_b128), then chain WMMAs.
        v16h aH[4];
#pragma unroll
        for (int kc = 0; kc < 4; ++kc)
            aH[kc] = load_A_lds(&sH[cur][0], HIDDEN, kc * 32, lane);

        v8f c = cE;
#pragma unroll
        for (int kc = 0; kc < 4; ++kc)
            c = __builtin_amdgcn_wmma_f32_16x16x32_f16(false, aH[kc], false, Bhh[kc], (short)0, c, false, false);

        // Gather next step's embedding A tile (no dependence on h) while c settles.
        v16h aEn = load_A_lds(&sEmb[idxN * EMBED], 0, 0, lane);

#pragma unroll
        for (int i = 0; i < 8; ++i) {
            const int m = i + mhigh;                              // C-layout row
            sH[nxt][m * HIDDEN + ncol + (lane & 15)] = (_Float16)tanhf(c[i]);
        }
        __syncthreads();   // h_new fully visible; also fences reads of sH[cur]

        // Next step's embedding contribution: off the h->h critical path.
        v8f cE2;
#pragma unroll
        for (int i = 0; i < 8; ++i) cE2[i] = biasH;
        cE2 = __builtin_amdgcn_wmma_f32_16x16x32_f16(false, aEn, false, Bih, (short)0, cE2, false, false);

        // ---- logits = h_new @ W_fc + b_fc ----
        v16h aL[4];
#pragma unroll
        for (int kc = 0; kc < 4; ++kc)
            aL[kc] = load_A_lds(&sH[nxt][0], HIDDEN, kc * 32, lane);

        v8f d;
#pragma unroll
        for (int i = 0; i < 8; ++i) d[i] = biasF;
#pragma unroll
        for (int kc = 0; kc < 4; ++kc)
            d = __builtin_amdgcn_wmma_f32_16x16x32_f16(false, aL[kc], false, Bfc[kc], (short)0, d, false, false);

#pragma unroll
        for (int i = 0; i < 8; ++i) {
            const int m = i + mhigh;
            out[(size_t)(rowBase + m) * (SEQ * VOCAB) + (size_t)t * VOCAB + ncol + (lane & 15)] = d[i];
        }

        cE = cE2;
        __syncthreads();   // all waves done reading sH[nxt] before t+1 overwrites sH[cur]
    }
}

extern "C" void kernel_launch(void* const* d_in, const int* in_sizes, int n_in,
                              void* d_out, int out_size, void* d_ws, size_t ws_size,
                              hipStream_t stream) {
    (void)in_sizes; (void)n_in; (void)out_size; (void)d_ws; (void)ws_size;
    const int*   x   = (const int*)d_in[0];
    const float* emb = (const float*)d_in[1];
    const float* Wih = (const float*)d_in[2];
    const float* Whh = (const float*)d_in[3];
    const float* bh  = (const float*)d_in[4];
    const float* Wfc = (const float*)d_in[5];
    const float* bfc = (const float*)d_in[6];
    float* out = (float*)d_out;

    dim3 grid(BATCH / TILE_M);          // 16 independent recurrence chains
    dim3 block(TILE_M * NWAVES * 2);    // 256 threads = 8 waves (wave32)
    hipLaunchKernelGGL(rnn_seq_kernel, grid, block, 0, stream,
                       x, emb, Wih, Whh, bh, Wfc, bfc, out);
}
